// GatedGraphConv_44667659878538
// MI455X (gfx1250) — compile-verified, matrix operand order checked by reading, not run on hardware
//
#include <hip/hip_runtime.h>
#include <math.h>

#define NNODE 50000
#define NEDGE 800000
#define INC   128
#define OUTC  64
#define SPLITN 812

typedef __attribute__((ext_vector_type(16))) _Float16 v16h;
typedef __attribute__((ext_vector_type(8)))  float    v8f;
typedef __attribute__((ext_vector_type(8)))  unsigned v8u;

// ---------------- utility kernels ----------------

__global__ void k_zero16(uint4* p, long n16) {
    long i = (long)blockIdx.x * blockDim.x + threadIdx.x;
    if (i < n16) p[i] = make_uint4(0u, 0u, 0u, 0u);
}

__global__ void k_cvt_f16(const float* __restrict__ s, _Float16* __restrict__ d, int n) {
    int i = blockIdx.x * blockDim.x + threadIdx.x;
    if (i < n) d[i] = (_Float16)s[i];
}

__global__ void k_degree(const int* __restrict__ col, float* __restrict__ deg, int E) {
    int i = blockIdx.x * blockDim.x + threadIdx.x;
    if (i < E)
        __hip_atomic_fetch_add(&deg[col[i]], 1.0f, __ATOMIC_RELAXED, __HIP_MEMORY_SCOPE_AGENT);
}

__global__ void k_rsq(float* __restrict__ deg, int n) {
    int i = blockIdx.x * blockDim.x + threadIdx.x;
    if (i < n) { float d = deg[i]; deg[i] = (d > 0.0f) ? rsqrtf(d) : 0.0f; }
}

// each edge handled by 32 lanes, 4 channels/lane (128 channels)
__global__ void k_scatter(const float* __restrict__ x, const int* __restrict__ row,
                          const int* __restrict__ col, const float* __restrict__ dis,
                          float* __restrict__ xhomo, float* __restrict__ xhete, int E) {
    long g = (long)blockIdx.x * blockDim.x + threadIdx.x;
    int e = (int)(g >> 5), lane = (int)(g & 31);
    if (e >= E) return;
    int r = row[e], c = col[e];
    float nrm = dis[r] * dis[c];
    bool homo = (r < SPLITN) == (c < SPLITN);
    float* dst = homo ? xhomo : xhete;
    const float4 v = *((const float4*)(x + (size_t)c * INC) + lane);
    float* op = dst + (size_t)r * INC + lane * 4;
    __hip_atomic_fetch_add(op + 0, v.x * nrm, __ATOMIC_RELAXED, __HIP_MEMORY_SCOPE_AGENT);
    __hip_atomic_fetch_add(op + 1, v.y * nrm, __ATOMIC_RELAXED, __HIP_MEMORY_SCOPE_AGENT);
    __hip_atomic_fetch_add(op + 2, v.z * nrm, __ATOMIC_RELAXED, __HIP_MEMORY_SCOPE_AGENT);
    __hip_atomic_fetch_add(op + 3, v.w * nrm, __ATOMIC_RELAXED, __HIP_MEMORY_SCOPE_AGENT);
}

// top-3 distinct-column hete neighbors via packed u64 atomicMax, 3 passes
__global__ void k_topk(const int* __restrict__ row, const int* __restrict__ col,
                       const float* __restrict__ dis, unsigned long long* __restrict__ slots,
                       int pass, int E) {
    int e = blockIdx.x * blockDim.x + threadIdx.x;
    if (e >= E) return;
    int r = row[e], c = col[e];
    if ((r < SPLITN) == (c < SPLITN)) return;              // hete only
    float nrm = dis[r] * dis[c];
    unsigned long long pk = ((unsigned long long)__float_as_uint(nrm) << 32)
                          | (unsigned long long)(0xFFFFFFFFu - (unsigned)c);
    if (pass >= 1) {
        unsigned long long s0 = slots[r];
        if ((0xFFFFFFFFu - (unsigned)(s0 & 0xFFFFFFFFull)) == (unsigned)c) return;
    }
    if (pass >= 2) {
        unsigned long long s1 = slots[NNODE + r];
        if ((0xFFFFFFFFu - (unsigned)(s1 & 0xFFFFFFFFull)) == (unsigned)c) return;
    }
    __hip_atomic_fetch_max(&slots[(size_t)pass * NNODE + r], pk,
                           __ATOMIC_RELAXED, __HIP_MEMORY_SCOPE_AGENT);
}

__global__ void k_nmean(const float* __restrict__ xe_t, const unsigned long long* __restrict__ slots,
                        float* __restrict__ nmean, float* __restrict__ cntf) {
    long g = (long)blockIdx.x * blockDim.x + threadIdx.x;
    int node = (int)(g >> 6), ch = (int)(g & 63);
    if (node >= NNODE) return;
    float s = 0.0f; int cnt = 0;
#pragma unroll
    for (int k = 0; k < 3; ++k) {
        unsigned long long sl = slots[(size_t)k * NNODE + node];
        if (sl != 0ull) {
            int c = (int)(0xFFFFFFFFu - (unsigned)(sl & 0xFFFFFFFFull));
            s += xe_t[(size_t)c * OUTC + ch];
            ++cnt;
        }
    }
    nmean[g] = cnt ? s / (float)cnt : 0.0f;
    if (ch == 0) cntf[node] = (float)cnt;
}

// ---------------- WMMA kernels ----------------
// 16-bit A-frag k index for half i of lane-half hi:  k = (i&7) + 16*(i>>3) + 8*hi
// B-frag gathered as 8 dwords (consecutive half pairs): kp = 2*(p&3) + 16*(p>>2) + 8*hi

__global__ __launch_bounds__(128) void k_linear_wmma(
        const float* __restrict__ xhomo, const float* __restrict__ xhete,
        const _Float16* __restrict__ wh, const _Float16* __restrict__ we,
        const float* __restrict__ bh, const float* __restrict__ be,
        float* __restrict__ out_xh /*d_out*/, float* __restrict__ out_xe /*xe_t*/) {
    __shared__ float sH[16 * 132];
    __shared__ float sE[16 * 132];
    int tile = blockIdx.x;
    size_t base = (size_t)tile * 16 * INC;
    for (int idx = threadIdx.x; idx < 16 * INC; idx += 128) {
        int r = idx >> 7, cc = idx & 127;
        sH[r * 132 + cc] = xhomo[base + idx];
        sE[r * 132 + cc] = xhete[base + idx];
    }
    __syncthreads();
    int lane = threadIdx.x & 31, wid = threadIdx.x >> 5;
    int m = lane & 15, hi = lane >> 4;
    int ob = wid << 4;
    int ocol = ob + m;                         // B-frag column (n == lane&15)
    const unsigned* whu = (const unsigned*)wh;
    const unsigned* weu = (const unsigned*)we;
    v8f cH = {}; v8f cE = {};
#pragma unroll
    for (int kk = 0; kk < INC; kk += 32) {
        v16h aH, aE;
#pragma unroll
        for (int i = 0; i < 16; ++i) {
            int kl = (i & 7) + ((i >> 3) << 4) + (hi << 3);
            aH[i] = (_Float16)sH[m * 132 + kk + kl];
            aE[i] = (_Float16)sE[m * 132 + kk + kl];
        }
        v8u buH, buE;
#pragma unroll
        for (int p = 0; p < 8; ++p) {
            int kp = 2 * (p & 3) + ((p >> 2) << 4) + (hi << 3);
            int di = (ocol * INC + kk + kp) >> 1;
            buH[p] = whu[di];
            buE[p] = weu[di];
        }
        cH = __builtin_amdgcn_wmma_f32_16x16x32_f16(false, aH, false,
                __builtin_bit_cast(v16h, buH), (short)0, cH, false, false);
        cE = __builtin_amdgcn_wmma_f32_16x16x32_f16(false, aE, false,
                __builtin_bit_cast(v16h, buE), (short)0, cE, false, false);
    }
    int n = lane & 15;
#pragma unroll
    for (int v = 0; v < 8; ++v) {
        int mm = v + (hi << 3);
        int node = tile * 16 + mm;
        int oc = ob + n;
        out_xh[(size_t)node * OUTC + oc] = cH[v] + bh[oc];
        out_xe[(size_t)node * OUTC + oc] = cE[v] + be[oc];
    }
}

// result[n,o] = sum_{d,e} self[n,d] * Wbil[o,d,e] * nmean[n,e]  as a GEMM over K=4096
// with P[n, d*64+e] = self[n,d]*nmean[n,e] built in registers. Epilogue does the
// full fuse:  out = xh_t + gate*leaky(bil) + (1-gate)*xe_t.
__global__ __launch_bounds__(128) void k_bilinear_wmma(
        const float* __restrict__ xe_t, const float* __restrict__ nmean,
        const float* __restrict__ cntf, const _Float16* __restrict__ wb,
        const float* __restrict__ bbil, const float* __restrict__ gw,
        float* __restrict__ out) {
    __shared__ float sS[16 * 68];
    __shared__ float sN[16 * 68];
    __shared__ float sC[16];
    int tile = blockIdx.x;
    size_t base = (size_t)tile * 16 * OUTC;
    for (int idx = threadIdx.x; idx < 16 * OUTC; idx += 128) {
        int r = idx >> 6, cc = idx & 63;
        sS[r * 68 + cc] = xe_t[base + idx];
        sN[r * 68 + cc] = nmean[base + idx];
    }
    if (threadIdx.x < 16) sC[threadIdx.x] = cntf[tile * 16 + threadIdx.x];
    __syncthreads();
    int lane = threadIdx.x & 31, wid = threadIdx.x >> 5;
    int m = lane & 15, hi = lane >> 4;
    int ob = wid << 4;
    int ocol = ob + m;
    // preload this lane's nmean halves for both 32-wide sub-blocks of e
    v16h nm0, nm1;
#pragma unroll
    for (int i = 0; i < 16; ++i) {
        int kl = (i & 7) + ((i >> 3) << 4) + (hi << 3);
        nm0[i] = (_Float16)sN[m * 68 + kl];
        nm1[i] = (_Float16)sN[m * 68 + 32 + kl];
    }
    const unsigned* wbu = (const unsigned*)wb;
    v8f c = {};
#pragma unroll 2
    for (int kk = 0; kk < 4096; kk += 32) {
        int d = kk >> 6;
        _Float16 s = (_Float16)sS[m * 68 + d];
        v16h a = ((kk & 32) ? nm1 : nm0) * s;        // P values: self[m,d]*nmean[m,e]
        v8u bu;
#pragma unroll
        for (int p = 0; p < 8; ++p) {
            int kp = 2 * (p & 3) + ((p >> 2) << 4) + (hi << 3);
            bu[p] = wbu[(ocol * 4096 + kk + kp) >> 1];
        }
        c = __builtin_amdgcn_wmma_f32_16x16x32_f16(false, a, false,
                __builtin_bit_cast(v16h, bu), (short)0, c, false, false);
    }
    int n = lane & 15;
#pragma unroll
    for (int v = 0; v < 8; ++v) {
        int mm = v + (hi << 3);
        int node = tile * 16 + mm;
        int oc = ob + n;
        float cnt = sC[mm];
        float y = (cnt > 0.0f) ? (c[v] + bbil[oc]) : 0.0f;
        float lr = (y > 0.0f) ? y : 0.01f * y;
        float g = 1.0f / (1.0f + expf(-gw[oc]));
        float fused = g * lr + (1.0f - g) * sS[mm * 68 + oc];
        size_t idx = (size_t)node * OUTC + oc;
        out[idx] = out[idx] + fused;                 // out already holds xh_t
    }
}

// ---------------- host launcher ----------------

extern "C" void kernel_launch(void* const* d_in, const int* in_sizes, int n_in,
                              void* d_out, int out_size, void* d_ws, size_t ws_size,
                              hipStream_t stream) {
    const float* x      = (const float*)d_in[0];
    const float* Whomo  = (const float*)d_in[1];
    const float* bhomo  = (const float*)d_in[2];
    const float* Whete  = (const float*)d_in[3];
    const float* bhete  = (const float*)d_in[4];
    const float* Wbil   = (const float*)d_in[5];
    const float* bbil   = (const float*)d_in[6];
    const float* gatew  = (const float*)d_in[7];
    const int*   erow   = (const int*)d_in[8];
    const int*   ecol   = erow + NEDGE;
    float* out = (float*)d_out;

    char* ws = (char*)d_ws;
    size_t p = 0;
    auto alloc = [&](size_t bytes) { size_t o = p; p += (bytes + 15) & ~(size_t)15; return o; };
    // zero-region must be contiguous from offset 0 (all sizes are multiples of 16)
    size_t off_deg   = alloc((size_t)NNODE * 4);                 // deg -> dis (in place)
    size_t off_slots = alloc((size_t)3 * NNODE * 8);             // top-3 packed slots
    size_t off_xh    = alloc((size_t)NNODE * INC * 4);           // x_homo
    size_t off_xe    = alloc((size_t)NNODE * INC * 4);           // x_hete
    size_t zbytes    = p;                                        // zero everything above
    size_t off_xet   = alloc((size_t)NNODE * OUTC * 4);          // xe_t
    size_t off_nm    = alloc((size_t)NNODE * OUTC * 4);          // mean neighbor
    size_t off_cnt   = alloc((size_t)NNODE * 4);                 // valid-neighbor count
    size_t off_whh   = alloc((size_t)OUTC * INC * 2);            // W_homo f16
    size_t off_weh   = alloc((size_t)OUTC * INC * 2);            // W_hete f16
    size_t off_wbh   = alloc((size_t)OUTC * OUTC * OUTC * 2);    // W_bil f16
    (void)ws_size; (void)in_sizes; (void)n_in; (void)out_size;

    float* deg   = (float*)(ws + off_deg);
    unsigned long long* slots = (unsigned long long*)(ws + off_slots);
    float* xhomo = (float*)(ws + off_xh);
    float* xhete = (float*)(ws + off_xe);
    float* xet   = (float*)(ws + off_xet);
    float* nm    = (float*)(ws + off_nm);
    float* cntf  = (float*)(ws + off_cnt);
    _Float16* whh = (_Float16*)(ws + off_whh);
    _Float16* weh = (_Float16*)(ws + off_weh);
    _Float16* wbh = (_Float16*)(ws + off_wbh);

    const int B = 256;
    long n16 = (long)(zbytes / 16);
    k_zero16<<<(int)((n16 + B - 1) / B), B, 0, stream>>>((uint4*)ws, n16);

    k_cvt_f16<<<(OUTC * INC + B - 1) / B, B, 0, stream>>>(Whomo, whh, OUTC * INC);
    k_cvt_f16<<<(OUTC * INC + B - 1) / B, B, 0, stream>>>(Whete, weh, OUTC * INC);
    k_cvt_f16<<<(OUTC * OUTC * OUTC + B - 1) / B, B, 0, stream>>>(Wbil, wbh, OUTC * OUTC * OUTC);

    k_degree<<<(NEDGE + B - 1) / B, B, 0, stream>>>(ecol, deg, NEDGE);
    k_rsq<<<(NNODE + B - 1) / B, B, 0, stream>>>(deg, NNODE);

    long sthreads = (long)NEDGE * 32;
    k_scatter<<<(int)((sthreads + B - 1) / B), B, 0, stream>>>(x, erow, ecol, deg,
                                                              xhomo, xhete, NEDGE);
    for (int pass = 0; pass < 3; ++pass)
        k_topk<<<(NEDGE + B - 1) / B, B, 0, stream>>>(erow, ecol, deg, slots, pass, NEDGE);

    int tiles = NNODE / 16;  // 3125, exact
    k_linear_wmma<<<tiles, 128, 0, stream>>>(xhomo, xhete, whh, weh, bhomo, bhete, out, xet);

    long mthreads = (long)NNODE * OUTC;
    k_nmean<<<(int)((mthreads + B - 1) / B), B, 0, stream>>>(xet, slots, nm, cntf);

    k_bilinear_wmma<<<tiles, 128, 0, stream>>>(xet, nm, cntf, wbh, bbil, gatew, out);
}
